// MoREModelSynthesisIOptionB_52888227283724
// MI455X (gfx1250) — matrix operand order repaired and checked by reading, not compiled, not run on hardware
//
#include <hip/hip_runtime.h>
#include <hip/hip_bf16.h>
#include <math.h>

// ---------------------------------------------------------------------------
// Model config (matches reference)
// ---------------------------------------------------------------------------
#define CB   2
#define CS   1024
#define CT   2048            // B*S tokens
#define CV   32000
#define CD   512
#define CH   8
#define CHD  64
#define CL   4
#define CE   8
#define CDFF_EXP 1024        // 2*D
#define CDFF_MLP 2048        // 4*D
#define CNR  2
#define CEPS 1e-6f

typedef __attribute__((ext_vector_type(16))) _Float16     v16h;
typedef __attribute__((ext_vector_type(8)))  float        v8f;
typedef __attribute__((ext_vector_type(4)))  unsigned int v4u;
typedef __attribute__((ext_vector_type(8)))  int          v8i;
typedef __attribute__((ext_vector_type(4)))  int          v4i;

// Tensor Data Mover availability (ROCm7.2 clang-22 = 5 args; clang-23 = 6 args)
#if defined(__has_builtin)
#  if __has_builtin(__builtin_amdgcn_tensor_load_to_lds) && __has_builtin(__builtin_amdgcn_s_wait_tensorcnt)
#    define HAVE_TDM 1
#  else
#    define HAVE_TDM 0
#  endif
#else
#  define HAVE_TDM 0
#endif

// ---------------------------------------------------------------------------
// Activation epilogues
// ---------------------------------------------------------------------------
template<int ACT>
__device__ __forceinline__ float act_f(float x) {
    if (ACT == 1) return 0.5f * x * (1.0f + erff(x * 0.70710678118f));     // exact GELU
    if (ACT == 2) return x / (1.0f + __expf(-x));                          // SiLU
    return x;
}

#if HAVE_TDM
// ---------------------------------------------------------------------------
// TDM: DMA a 2-D f32 panel (up to 32 rows x 64 cols) from global into LDS.
// tensor_dim0/1 give the in-bounds extent; DMA zero-fills OOB reads, which
// replaces manual edge padding. Tracked on TENSORcnt.
// ---------------------------------------------------------------------------
__device__ __forceinline__ void tdm_load_b_panel(const float* gsrc, unsigned ldsByteAddr,
                                                 unsigned cols, unsigned rows, unsigned ldbElems)
{
    unsigned long long ga = (unsigned long long)(const void*)gsrc;
    v4u g0;
    g0[0] = 1u;                                                    // D# count = 1 (user mode)
    g0[1] = ldsByteAddr;                                           // lds_addr
    g0[2] = (unsigned)(ga & 0xffffffffull);                        // global_addr lo
    g0[3] = (unsigned)((ga >> 32) & 0x01ffffffull) | 0x80000000u;  // global_addr hi | type=2
    v8i g1;
    g1[0] = (int)(2u << 16);                                       // data_size = 4 bytes
    g1[1] = (int)((cols & 0xffffu) << 16);                         // tensor_dim0 lo16 @ [63:48]
    g1[2] = (int)((cols >> 16) | ((rows & 0xffffu) << 16));        // dim0 hi16 | dim1 lo16
    g1[3] = (int)((rows >> 16) | (64u << 16));                     // dim1 hi16 | tile_dim0=64
    g1[4] = (int)32u;                                              // tile_dim1=32, tile_dim2=0
    g1[5] = (int)ldbElems;                                         // tensor_dim0_stride lo32
    g1[6] = 0;                                                     // stride0 hi16 | stride1 lo16
    g1[7] = 0;
    v4i gz = {0, 0, 0, 0};
#if __clang_major__ >= 23
    v8i gz8 = {0, 0, 0, 0, 0, 0, 0, 0};
    __builtin_amdgcn_tensor_load_to_lds(g0, g1, gz, gz, gz8, 0);
#else
    __builtin_amdgcn_tensor_load_to_lds(g0, g1, gz, gz, 0);
#endif
}
#endif

// ---------------------------------------------------------------------------
// Generic WMMA GEMM: C[M,N] = act(A[M,K] @ B[K,N])
//  - A staged to LDS pre-swizzled into WMMA A-fragment order (contiguous
//    32B per lane -> ds_load_b128 fragment loads).
//  - B panel DMA'd by the Tensor Data Mover into a double-buffered f32 LDS
//    tile (next panel in flight during current tile's WMMAs); converted to
//    f16 inline while building fragments. Fallback: manual staging.
//  - Optional row gather for A / row scatter for C + per-z expert segments.
// ---------------------------------------------------------------------------
template<int ACT>
__global__ __launch_bounds__(256)
void gemm_f16_wmma(const float* __restrict__ A, const float* __restrict__ Bw,
                   float* __restrict__ C,
                   int M, int N, int K, int lda, int ldb, int ldc,
                   const int* __restrict__ a_gather,
                   const int* __restrict__ c_scatter,
                   const int* __restrict__ seg_cnt,
                   const int* __restrict__ seg_off,
                   long bStrideZ)
{
    const int tid  = threadIdx.x;
    const int wave = tid >> 5;
    const int lane = tid & 31;
    const int z    = blockIdx.z;

    int Mz = M, segBase = 0;
    if (seg_cnt) { Mz = seg_cnt[z]; segBase = seg_off[z]; }
    const int m0 = blockIdx.y * 64;
    const int n0 = blockIdx.x * 64;
    if (m0 >= Mz) return;                       // uniform per block

    const float* __restrict__ B = Bw + (long)z * bStrideZ;

    __shared__ alignas(32) _Float16 sAswz[4 * 32 * 16];   // 4 row-subtiles, frag order
#if HAVE_TDM
    __shared__ alignas(16) float sBf32[2][32 * 64];       // TDM double buffer
#else
    __shared__ alignas(16) float sBf32[1][32 * 64];
#endif

    const int wm = (wave & 3) * 16;             // wave row offset in block tile
    const int wn = (wave >> 2) * 32;            // wave col offset in block tile
    const int hi = lane >> 4;
    const int ml = lane & 15;

    v8f c0 = {}; v8f c1 = {};

#if HAVE_TDM
    if (wave == 0)
        tdm_load_b_panel(B + n0, (unsigned)(size_t)(const void*)&sBf32[0][0],
                         (unsigned)(N - n0), (unsigned)K, (unsigned)ldb);
#endif

    for (int k0 = 0; k0 < K; k0 += 32) {
#if HAVE_TDM
        const int buf = (k0 >> 5) & 1;
#else
        const int buf = 0;
#endif
        // ---- stage A 64x32, pre-swizzled into fragment order ----
        #pragma unroll
        for (int s = 0; s < 8; ++s) {
            int i  = tid + s * 256;
            int r  = i >> 5, kk = i & 31;
            int lm = m0 + r;
            float v = 0.f;
            if (lm < Mz && (k0 + kk) < K) {
                int row = a_gather ? a_gather[segBase + lm] : lm;
                v = A[(long)row * lda + k0 + kk];
            }
            // dest: subtile r>>4, lane = ((kk>>3)&1)*16 + (r&15), elem = (kk>>4)*8 + (kk&7)
            int dl = ((kk >> 3) & 1) * 16 + (r & 15);
            int de = ((kk >> 4) << 3) + (kk & 7);
            sAswz[(((r >> 4) * 32 + dl) << 4) + de] = (_Float16)v;
        }
        // prefetch next A panel rows into cache
        if (k0 + 32 < K && tid < 64) {
            int lm = m0 + tid;
            if (lm < Mz) {
                int row = a_gather ? a_gather[segBase + lm] : lm;
                __builtin_prefetch(&A[(long)row * lda + k0 + 32], 0, 1);
            }
        }
#if HAVE_TDM
        if (wave == 0) {
            if (k0 + 32 < K) {
                tdm_load_b_panel(B + (long)(k0 + 32) * ldb + n0,
                                 (unsigned)(size_t)(const void*)&sBf32[buf ^ 1][0],
                                 (unsigned)(N - n0), (unsigned)(K - (k0 + 32)), (unsigned)ldb);
                __builtin_amdgcn_s_wait_tensorcnt(1);   // current panel landed
            } else {
                __builtin_amdgcn_s_wait_tensorcnt(0);
            }
        }
#else
        #pragma unroll
        for (int s = 0; s < 8; ++s) {
            int i  = tid + s * 256;
            int kk = i >> 6, cc = i & 63;
            float v = 0.f;
            if ((k0 + kk) < K && (n0 + cc) < N)
                v = B[(long)(k0 + kk) * ldb + n0 + cc];
            sBf32[0][kk * 64 + cc] = v;
        }
#endif
        __syncthreads();

        // ---- fragments ----
        const v16h af = *(const v16h*)&sAswz[(((wave & 3) * 32 + lane) << 4)];
        v16h bf0, bf1;
        #pragma unroll
        for (int i = 0; i < 16; ++i) {
            int kk = ((i >> 3) << 4) + hi * 8 + (i & 7);
            bf0[i] = (_Float16)sBf32[buf][kk * 64 + wn + ml];
            bf1[i] = (_Float16)sBf32[buf][kk * 64 + wn + 16 + ml];
        }
        c0 = __builtin_amdgcn_wmma_f32_16x16x32_f16(false, af, false, bf0, (short)0, c0, false, false);
        c1 = __builtin_amdgcn_wmma_f32_16x16x32_f16(false, af, false, bf1, (short)0, c1, false, false);
        __syncthreads();
    }

    // ---- epilogue + store (C layout: vgpr j -> row j + hi*8, col = lane&15) ----
    #pragma unroll
    for (int j = 0; j < 8; ++j) {
        int lm = m0 + wm + hi * 8 + j;
        if (lm >= Mz) continue;
        long crow = c_scatter ? (long)c_scatter[segBase + lm] : (long)lm;
        int cc0 = n0 + wn + ml;
        int cc1 = cc0 + 16;
        if (cc0 < N) C[crow * ldc + cc0] = act_f<ACT>(c0[j]);
        if (cc1 < N) C[crow * ldc + cc1] = act_f<ACT>(c1[j]);
    }
}

// ---------------------------------------------------------------------------
// Flash attention with WMMA (causal, H=8, HD=64).
// grid: (S/128, B*H); block 256 = 8 waves; each wave: one 16-row query tile.
// K/V/P staged in LDS pre-swizzled into WMMA fragment order (b128 loads).
// ---------------------------------------------------------------------------
__global__ __launch_bounds__(256)
void attn_flash_wmma(const float* __restrict__ Q, const float* __restrict__ Km,
                     const float* __restrict__ Vm, float* __restrict__ O)
{
    const int tid  = threadIdx.x;
    const int wave = tid >> 5;
    const int lane = tid & 31;
    const int hi   = lane >> 4;
    const int ml   = lane & 15;

    const int bh = blockIdx.y;
    const int b  = bh >> 3, h = bh & 7;
    const int qblock = blockIdx.x * 128;
    const int qrow0  = qblock + wave * 16;
    const long baseT = (long)b * CS;
    const int  hc    = h * CHD;

    // frag-order buffers: [frag][lane][16 halves]
    __shared__ alignas(32) _Float16 sKswz[4 * 32 * 16];   // (dc*2+kt) frags of K^T
    __shared__ alignas(32) _Float16 sVswz[4 * 32 * 16];   // 4 n-group frags of V
    __shared__ alignas(32) _Float16 sPswz[8][32 * 16];    // per-wave P in A-frag order

    // ---- load Q fragments once (scale 1/sqrt(64) folded in) ----
    v16h qa0, qa1;
    {
        const int qr = qrow0 + ml;
        const float* qp = Q + (baseT + qr) * CD + hc;
        #pragma unroll
        for (int i = 0; i < 16; ++i) {
            int r = i >> 1, p = i & 1;
            int kk = hi * 16 + r * 2 + p;
            qa0[i] = (_Float16)(qp[kk] * 0.125f);
            qa1[i] = (_Float16)(qp[32 + kk] * 0.125f);
        }
    }

    v8f o0 = {}, o1 = {}, o2 = {}, o3 = {};
    float mi[8], li[8];
    #pragma unroll
    for (int j = 0; j < 8; ++j) { mi[j] = -1e30f; li[j] = 0.f; }

    const int kend = qblock + 127;              // causal bound for the whole block
    for (int kbase = 0; kbase <= kend; kbase += 32) {
        // ---- stage K,V chunk (32 keys x 64 dims), swizzled to frag order ----
        #pragma unroll
        for (int s = 0; s < 8; ++s) {
            int i  = tid + s * 256;
            int kk = i >> 6, d = i & 63;
            float kvK = Km[(baseT + kbase + kk) * CD + hc + d];
            float kvV = Vm[(baseT + kbase + kk) * CD + hc + d];
            // K^T frag (kt=kk>>4, dc=d>>5): lane=((d>>4)&1)*16+(kk&15), elem=d&15
            {
                int fr = ((d >> 5) << 1) + (kk >> 4);
                int dl = ((d >> 4) & 1) * 16 + (kk & 15);
                sKswz[((fr * 32 + dl) << 4) + (d & 15)] = (_Float16)kvK;
            }
            // V frag (g=d>>4): lane=((kk>>3)&1)*16+(d&15), elem=(kk>>4)*8+(kk&7)
            {
                int fr = d >> 4;
                int dl = (((kk >> 3) & 1) << 4) + (d & 15);
                sVswz[((fr * 32 + dl) << 4) + ((kk >> 4) << 3) + (kk & 7)] = (_Float16)kvV;
            }
        }
        __syncthreads();

        // ---- scores: S(16x32) = Q(16x64) @ K^T ----
        v8f s0 = {}, s1 = {};
        {
            const v16h bk00 = *(const v16h*)&sKswz[((0 * 32 + lane) << 4)];  // kt0, dc0
            const v16h bk01 = *(const v16h*)&sKswz[((2 * 32 + lane) << 4)];  // kt0, dc1
            const v16h bk10 = *(const v16h*)&sKswz[((1 * 32 + lane) << 4)];  // kt1, dc0
            const v16h bk11 = *(const v16h*)&sKswz[((3 * 32 + lane) << 4)];  // kt1, dc1
            s0 = __builtin_amdgcn_wmma_f32_16x16x32_f16(false, qa0, false, bk00, (short)0, s0, false, false);
            s0 = __builtin_amdgcn_wmma_f32_16x16x32_f16(false, qa1, false, bk01, (short)0, s0, false, false);
            s1 = __builtin_amdgcn_wmma_f32_16x16x32_f16(false, qa0, false, bk10, (short)0, s1, false, false);
            s1 = __builtin_amdgcn_wmma_f32_16x16x32_f16(false, qa1, false, bk11, (short)0, s1, false, false);
        }

        // ---- online softmax per row; write P directly in A-frag order ----
        const int key0 = kbase + ml, key1 = kbase + 16 + ml;
        #pragma unroll
        for (int j = 0; j < 8; ++j) {
            const int row = qrow0 + hi * 8 + j;
            const bool ok0 = (key0 <= row), ok1 = (key1 <= row);
            float a0 = ok0 ? s0[j] : -1e30f;
            float a1 = ok1 ? s1[j] : -1e30f;
            float rmax = fmaxf(a0, a1);
            rmax = fmaxf(rmax, __shfl_xor(rmax, 1, 32));
            rmax = fmaxf(rmax, __shfl_xor(rmax, 2, 32));
            rmax = fmaxf(rmax, __shfl_xor(rmax, 4, 32));
            rmax = fmaxf(rmax, __shfl_xor(rmax, 8, 32));
            float nm    = fmaxf(mi[j], rmax);
            float alpha = __expf(mi[j] - nm);
            float p0 = ok0 ? __expf(a0 - nm) : 0.f;
            float p1 = ok1 ? __expf(a1 - nm) : 0.f;
            float ps = p0 + p1;
            ps += __shfl_xor(ps, 1, 32);
            ps += __shfl_xor(ps, 2, 32);
            ps += __shfl_xor(ps, 4, 32);
            ps += __shfl_xor(ps, 8, 32);
            li[j] = li[j] * alpha + ps;
            mi[j] = nm;
            o0[j] *= alpha; o1[j] *= alpha; o2[j] *= alpha; o3[j] *= alpha;
            // A-frag dest for (m=hi*8+j, kk=ml / 16+ml):
            //   lane=(ml>>3)*16 + m, elem = (ml&7) (+8 for the second key tile)
            const int dl = ((ml >> 3) << 4) + hi * 8 + j;
            sPswz[wave][(dl << 4) + (ml & 7)]     = (_Float16)p0;
            sPswz[wave][(dl << 4) + 8 + (ml & 7)] = (_Float16)p1;
        }

        // ---- O(16x64) += P(16x32) @ V(32x64) ----
        {
            const v16h pa  = *(const v16h*)&sPswz[wave][(lane << 4)];
            const v16h bv0 = *(const v16h*)&sVswz[((0 * 32 + lane) << 4)];
            const v16h bv1 = *(const v16h*)&sVswz[((1 * 32 + lane) << 4)];
            const v16h bv2 = *(const v16h*)&sVswz[((2 * 32 + lane) << 4)];
            const v16h bv3 = *(const v16h*)&sVswz[((3 * 32 + lane) << 4)];
            o0 = __builtin_amdgcn_wmma_f32_16x16x32_f16(false, pa, false, bv0, (short)0, o0, false, false);
            o1 = __builtin_amdgcn_wmma_f32_16x16x32_f16(false, pa, false, bv1, (short)0, o1, false, false);
            o2 = __builtin_amdgcn_wmma_f32_16x16x32_f16(false, pa, false, bv2, (short)0, o2, false, false);
            o3 = __builtin_amdgcn_wmma_f32_16x16x32_f16(false, pa, false, bv3, (short)0, o3, false, false);
        }
        __syncthreads();
    }

    // ---- normalize + store ----
    #pragma unroll
    for (int j = 0; j < 8; ++j) {
        const int row = qrow0 + hi * 8 + j;
        const float inv = 1.f / li[j];
        float* op = O + (baseT + row) * CD + hc;
        op[ml]      = o0[j] * inv;
        op[16 + ml] = o1[j] * inv;
        op[32 + ml] = o2[j] * inv;
        op[48 + ml] = o3[j] * inv;
    }
}

// ---------------------------------------------------------------------------
// Elementwise / small kernels
// ---------------------------------------------------------------------------
__global__ void embed_k(const int* __restrict__ ids, const float* __restrict__ tok,
                        const float* __restrict__ pos, float* __restrict__ x)
{
    const long t = blockIdx.x;
    const int  id = ids[t];
    const int  s  = (int)(t % CS);
    for (int i = threadIdx.x; i < CD; i += blockDim.x)
        x[t * CD + i] = tok[(long)id * CD + i] + pos[(long)s * CD + i];
}

__global__ void rmsnorm_k(const float* __restrict__ x, const float* __restrict__ w,
                          float* __restrict__ y)
{
    const long row = blockIdx.x;
    const int  tid = threadIdx.x;
    const float* xr = x + row * CD;
    float s = 0.f;
    for (int i = tid; i < CD; i += blockDim.x) { float v = xr[i]; s += v * v; }
    s += __shfl_xor(s, 1, 32);  s += __shfl_xor(s, 2, 32);
    s += __shfl_xor(s, 4, 32);  s += __shfl_xor(s, 8, 32);  s += __shfl_xor(s, 16, 32);
    __shared__ float red[8];
    if ((tid & 31) == 0) red[tid >> 5] = s;
    __syncthreads();
    if (tid < 32) {
        float v = (tid < 8) ? red[tid] : 0.f;
        v += __shfl_xor(v, 1, 32); v += __shfl_xor(v, 2, 32); v += __shfl_xor(v, 4, 32);
        if (tid == 0) red[0] = v;
    }
    __syncthreads();
    const float inv = rsqrtf(red[0] * (1.0f / CD) + CEPS);
    for (int i = tid; i < CD; i += blockDim.x)
        y[row * CD + i] = xr[i] * inv * w[i];
}

__global__ void add_k(float* __restrict__ a, const float* __restrict__ b, long n)
{
    long i = (long)blockIdx.x * blockDim.x + threadIdx.x;
    if (i < n) a[i] += b[i];
}

__global__ void concat_k(const float* __restrict__ a, const float* __restrict__ b,
                         float* __restrict__ c)
{
    const long t = blockIdx.x;
    for (int i = threadIdx.x; i < CD; i += blockDim.x) {
        c[t * (2 * CD) + i]      = a[t * CD + i];
        c[t * (2 * CD) + CD + i] = b[t * CD + i];
    }
}

__device__ __forceinline__ unsigned wang_h(unsigned s)
{
    s = (s ^ 61u) ^ (s >> 16); s *= 9u; s ^= s >> 4; s *= 0x27d4eb2du; s ^= s >> 15;
    return s;
}

__global__ void stateinit_k(float* __restrict__ st, int layer)
{
    const long t = blockIdx.x;
    for (int i = threadIdx.x; i < CD; i += blockDim.x) {
        unsigned h1 = wang_h((unsigned)(layer * 1315423911u) ^ (unsigned)((t * CD + i) * 2654435761u));
        unsigned h2 = wang_h(h1 ^ 0x9e3779b9u);
        float u1 = ((h1 >> 8) + 1) * 5.9604645e-8f;
        float u2 = (h2 >> 8) * 5.9604645e-8f;
        st[t * CD + i] = 0.02f * sqrtf(-2.f * logf(u1)) * cosf(6.2831853f * u2);
    }
}

__global__ void routerzero_k(int* counts, float* facc, float* pacc)
{
    int e = threadIdx.x;
    if (e < CE) { counts[e] = 0; facc[e] = 0.f; pacc[e] = 0.f; }
}

__global__ void router_k(const float* __restrict__ logits, int* __restrict__ eid,
                         int* __restrict__ counts, float* __restrict__ facc,
                         float* __restrict__ pacc)
{
    int t = blockIdx.x * blockDim.x + threadIdx.x;
    if (t >= CT) return;
    float v[CE]; float mx = -1e30f; int am = 0;
    #pragma unroll
    for (int e = 0; e < CE; ++e) { v[e] = logits[(long)t * CE + e]; if (v[e] > mx) { mx = v[e]; am = e; } }
    float se = 0.f;
    #pragma unroll
    for (int e = 0; e < CE; ++e) { v[e] = __expf(v[e] - mx); se += v[e]; }
    const float invs = 1.f / (se * (float)CT);
    #pragma unroll
    for (int e = 0; e < CE; ++e) atomicAdd(&pacc[e], v[e] * invs);
    eid[t] = am;
    atomicAdd(&counts[am], 1);
    atomicAdd(&facc[am], 1.f / (float)CT);
}

__global__ void scan_k(const int* __restrict__ counts, int* __restrict__ offs,
                       int* __restrict__ cursor)
{
    if (threadIdx.x == 0 && blockIdx.x == 0) {
        int s = 0;
        for (int e = 0; e < CE; ++e) { offs[e] = s; cursor[e] = s; s += counts[e]; }
    }
}

__global__ void scatter_k(const int* __restrict__ eid, int* __restrict__ cursor,
                          int* __restrict__ idx)
{
    int t = blockIdx.x * blockDim.x + threadIdx.x;
    if (t >= CT) return;
    int p = atomicAdd(&cursor[eid[t]], 1);
    idx[p] = t;
}

__global__ void lb_k(const float* __restrict__ facc, const float* __restrict__ pacc,
                     float* __restrict__ aux)
{
    if (threadIdx.x == 0 && blockIdx.x == 0) {
        float s = 0.f;
        for (int e = 0; e < CE; ++e) s += facc[e] * pacc[e];
        *aux += (float)CE * s;
    }
}

__global__ void auxzero_k(float* aux) { if (threadIdx.x == 0) *aux = 0.f; }

// ---------------------------------------------------------------------------
// Host-side helpers
// ---------------------------------------------------------------------------
static void launch_gemm(hipStream_t st, int act,
                        const float* A, const float* B, float* C,
                        int M, int N, int K, int lda, int ldb, int ldc,
                        const int* ag = nullptr, const int* cs = nullptr,
                        const int* segc = nullptr, const int* sego = nullptr,
                        int Z = 1, long bstride = 0)
{
    dim3 g((N + 63) / 64, (M + 63) / 64, Z), b(256);
    if (act == 1)
        gemm_f16_wmma<1><<<g, b, 0, st>>>(A, B, C, M, N, K, lda, ldb, ldc, ag, cs, segc, sego, bstride);
    else if (act == 2)
        gemm_f16_wmma<2><<<g, b, 0, st>>>(A, B, C, M, N, K, lda, ldb, ldc, ag, cs, segc, sego, bstride);
    else
        gemm_f16_wmma<0><<<g, b, 0, st>>>(A, B, C, M, N, K, lda, ldb, ldc, ag, cs, segc, sego, bstride);
}

struct LayerP {
    const float *norm1, *norm2, *wq, *wk, *wv, *wo, *router, *ew1, *ew2;
    const float *adapter, *n1, *n2, *n3, *n4, *s_wq, *s_wk, *s_wv, *s_wo, *mlp1, *mlp2;
};

extern "C" void kernel_launch(void* const* d_in, const int* in_sizes, int n_in,
                              void* d_out, int out_size, void* d_ws, size_t ws_size,
                              hipStream_t stream)
{
    (void)in_sizes; (void)n_in; (void)ws_size;
    // ---- inputs (setup_inputs dict order, recursively flattened) ----
    const int*   ids     = (const int*)d_in[0];
    const float* tok_emb = (const float*)d_in[1];
    const float* pos_emb = (const float*)d_in[2];
    LayerP L[CL];
    int p = 3;
    for (int l = 0; l < CL; ++l) {
        L[l].norm1   = (const float*)d_in[p++];
        L[l].norm2   = (const float*)d_in[p++];
        L[l].wq      = (const float*)d_in[p++];
        L[l].wk      = (const float*)d_in[p++];
        L[l].wv      = (const float*)d_in[p++];
        L[l].wo      = (const float*)d_in[p++];
        L[l].router  = (const float*)d_in[p++];
        L[l].ew1     = (const float*)d_in[p++];
        L[l].ew2     = (const float*)d_in[p++];
        L[l].adapter = (const float*)d_in[p++];
        L[l].n1      = (const float*)d_in[p++];
        L[l].n2      = (const float*)d_in[p++];
        L[l].n3      = (const float*)d_in[p++];
        L[l].n4      = (const float*)d_in[p++];
        L[l].s_wq    = (const float*)d_in[p++];
        L[l].s_wk    = (const float*)d_in[p++];
        L[l].s_wv    = (const float*)d_in[p++];
        L[l].s_wo    = (const float*)d_in[p++];
        L[l].mlp1    = (const float*)d_in[p++];
        L[l].mlp2    = (const float*)d_in[p++];
    }
    const float* norm_out = (const float*)d_in[p++];
    const float* lm_head  = (const float*)d_in[p++];

    // ---- workspace carve ----
    float* F = (float*)d_ws;
    long off = 0;
    auto alloc = [&](long n) { float* q = F + off; off += n; return q; };
    float* x     = alloc((long)CT * CD);
    float* xn    = alloc((long)CT * CD);
    float* qb    = alloc((long)CT * CD);
    float* kb    = alloc((long)CT * CD);
    float* vb    = alloc((long)CT * CD);
    float* ctx   = alloc((long)CT * CD);
    float* tmp   = alloc((long)CT * CD);     // attn proj / expert_out (xf of SR)
    float* state = alloc((long)CT * CD);
    float* cat   = alloc((long)CT * 2 * CD);
    float* t0    = alloc((long)CT * CDFF_MLP);
    float* probs = alloc((long)CT * CE);
    float* facc  = alloc(CE);
    float* pacc  = alloc(CE);
    int* ib     = (int*)(F + off);
    int* eid    = ib;                 // [CT]
    int* idxl   = ib + CT;            // [CT]
    int* counts = ib + 2 * CT;        // [CE]
    int* offs   = counts + CE;        // [CE]
    int* cursor = offs + CE;          // [CE]

    float* logits = (float*)d_out;
    float* aux    = logits + ((long)out_size - 1);

    const long NTD = (long)CT * CD;
    dim3 b256(256);
    dim3 gRow(CT), gTok((CT + 255) / 256), gAdd((unsigned)((NTD + 255) / 256));

    auxzero_k<<<1, 32, 0, stream>>>(aux);

    // ---- embedding ----
    embed_k<<<gRow, b256, 0, stream>>>(ids, tok_emb, pos_emb, x);

    for (int l = 0; l < CL; ++l) {
        const LayerP& P = L[l];
        // ---- attention block ----
        rmsnorm_k<<<gRow, b256, 0, stream>>>(x, P.norm1, xn);
        launch_gemm(stream, 0, xn, P.wq, qb, CT, CD, CD, CD, CD, CD);
        launch_gemm(stream, 0, xn, P.wk, kb, CT, CD, CD, CD, CD, CD);
        launch_gemm(stream, 0, xn, P.wv, vb, CT, CD, CD, CD, CD, CD);
        attn_flash_wmma<<<dim3(CS / 128, CB * CH), b256, 0, stream>>>(qb, kb, vb, ctx);
        launch_gemm(stream, 0, ctx, P.wo, tmp, CT, CD, CD, CD, CD, CD);
        add_k<<<gAdd, b256, 0, stream>>>(x, tmp, NTD);

        // ---- MoE: route (top-1) + grouped expert GEMMs ----
        rmsnorm_k<<<gRow, b256, 0, stream>>>(x, P.norm2, xn);
        launch_gemm(stream, 0, xn, P.router, probs, CT, CE, CD, CD, CE, CE);
        routerzero_k<<<1, 32, 0, stream>>>(counts, facc, pacc);
        router_k<<<gTok, b256, 0, stream>>>(probs, eid, counts, facc, pacc);
        scan_k<<<1, 32, 0, stream>>>(counts, offs, cursor);
        scatter_k<<<gTok, b256, 0, stream>>>(eid, cursor, idxl);
        // h = gelu(x_e @ w1_e): gather by token list, scatter hidden back by token id
        launch_gemm(stream, 1, xn, P.ew1, t0, CT, CDFF_EXP, CD, CD, CDFF_EXP, CDFF_EXP,
                    idxl, idxl, counts, offs, CE, (long)CD * CDFF_EXP);
        // y = h @ w2_e  (combine weight == 1.0 for top-1)
        launch_gemm(stream, 0, t0, P.ew2, tmp, CT, CD, CDFF_EXP, CDFF_EXP, CD, CD,
                    idxl, idxl, counts, offs, CE, (long)CDFF_EXP * CD);
        lb_k<<<1, 32, 0, stream>>>(facc, pacc, aux);

        // ---- shared recurrent refinement (NR=2), xf := tmp (expert_out) ----
        stateinit_k<<<gRow, b256, 0, stream>>>(state, l);
        for (int r = 0; r < CNR; ++r) {
            concat_k<<<gRow, b256, 0, stream>>>(state, tmp, cat);
            launch_gemm(stream, 0, cat, P.adapter, state, CT, CD, 2 * CD, 2 * CD, CD, CD);
            // attn over seq_len==1 reduces to (rms(state) @ wv) @ wo
            rmsnorm_k<<<gRow, b256, 0, stream>>>(state, P.n1, qb);
            launch_gemm(stream, 0, qb, P.s_wv, kb, CT, CD, CD, CD, CD, CD);
            launch_gemm(stream, 0, kb, P.s_wo, qb, CT, CD, CD, CD, CD, CD);
            add_k<<<gAdd, b256, 0, stream>>>(state, qb, NTD);
            rmsnorm_k<<<gRow, b256, 0, stream>>>(state, P.n2, state);
            // MLP
            rmsnorm_k<<<gRow, b256, 0, stream>>>(state, P.n3, qb);
            launch_gemm(stream, 2, qb, P.mlp1, t0, CT, CDFF_MLP, CD, CD, CDFF_MLP, CDFF_MLP);
            launch_gemm(stream, 0, t0, P.mlp2, kb, CT, CD, CDFF_MLP, CDFF_MLP, CD, CD);
            add_k<<<gAdd, b256, 0, stream>>>(state, kb, NTD);
            rmsnorm_k<<<gRow, b256, 0, stream>>>(state, P.n4, state);
        }
        add_k<<<gAdd, b256, 0, stream>>>(x, state, NTD);
    }

    // ---- final norm + lm_head ----
    rmsnorm_k<<<gRow, b256, 0, stream>>>(x, norm_out, xn);
    launch_gemm(stream, 0, xn, lm_head, logits, CT, CV, CD, CD, CV, CV);
}
